// M3GNet_48043504173400
// MI455X (gfx1250) — matrix operand order repaired and checked by reading, not compiled, not run on hardware
//
#include <hip/hip_runtime.h>
#include <hip/hip_bf16.h>

#define N_NODES     20000
#define N_EDGES     640000
#define N_TRIPLETS  1000000
#define N_GRAPHS    64
#define N_RADIAL    36
#define UNITS       64
#define PI_F        3.14159265358979323846f
#define INV_CUTOFF  0.2f

typedef __attribute__((ext_vector_type(16))) _Float16 v16h;
typedef __attribute__((ext_vector_type(8)))  _Float16 v8h;
typedef __attribute__((ext_vector_type(4)))  _Float16 v4h;
typedef __attribute__((ext_vector_type(8)))  float    v8f;

union h16frag { v16h v; v8h h[2]; };

__device__ __forceinline__ float sgm(float x) {
    // fast sigmoid: v_exp_f32 + v_add + v_rcp_f32 (avoid IEEE div expansion)
    return __builtin_amdgcn_rcpf(1.0f + __expf(-x));
}
__device__ __forceinline__ float silu_f(float x) { return x * sgm(x); }

__device__ __forceinline__ void atomAddF(float* p, float v) {
    __hip_atomic_fetch_add(p, v, __ATOMIC_RELAXED, __HIP_MEMORY_SCOPE_AGENT);
}

// ---------------------------------------------------------------------------
// WMMA fragment helpers (layouts per CDNA5 ISA 7.12.2, wave32).
// A 16x32 f16 from row-major LDS (ld halves): per lane two contiguous
// 16-byte runs -> two ds_load_b128.
__device__ __forceinline__ v16h load_a_frag(const _Float16* base, int ld) {
    int lane = threadIdx.x & 31;
    const _Float16* row = base + (lane & 15) * ld + ((lane >> 4) << 3);
    h16frag f;
    f.h[0] = *(const v8h*)(row);
    f.h[1] = *(const v8h*)(row + 16);
    return f.v;
}

// B 32x16 f16 from N-major (transposed) LDS weights (ldk halves per N row):
// per lane two contiguous 8-half runs (k = 8h.., 16+8h..) -> two ds_load_b128.
// WTchunk points at [n_tile_base row][k chunk 0].
__device__ __forceinline__ v16h load_bT_frag(const _Float16* WTchunk, int ldk) {
    int lane = threadIdx.x & 31;
    const _Float16* p = WTchunk + (lane & 15) * ldk + ((lane >> 4) << 3);
    h16frag f;
    f.h[0] = *(const v8h*)(p);
    f.h[1] = *(const v8h*)(p + 16);
    return f.v;
}

__device__ __forceinline__ v8f wmma_f16(v16h a, v16h b, v8f c) {
    return __builtin_amdgcn_wmma_f32_16x16x32_f16(false, a, false, b, (short)0, c, false, false);
}

// D fragment coords: VGPR r -> M = r + (lane>=16 ? 8 : 0), N = lane & 15.

// ---------------------------------------------------------------------------
// K0: convert weights to f16, TRANSPOSED to N-major (W_tb1 zero-padded K 12->32).
__global__ __launch_bounds__(256) void convert_weights_kernel(
    const float* __restrict__ Wtb1, const float* __restrict__ Wtb2,
    const float* __restrict__ Wega, const float* __restrict__ Wegg,
    const float* __restrict__ Wata, const float* __restrict__ Watg,
    _Float16* __restrict__ o_tb1, _Float16* __restrict__ o_tb2,
    _Float16* __restrict__ o_ega, _Float16* __restrict__ o_egg,
    _Float16* __restrict__ o_ata, _Float16* __restrict__ o_atg)
{
    int t = blockIdx.x * 256 + threadIdx.x;
    if (t < 3 * 64 * 32) {               // [blk][n=64][k=32]
        int blk = t / (64 * 32);
        int rc  = t % (64 * 32);
        int n   = rc >> 5, k = rc & 31;
        float v = (k < 12) ? Wtb1[blk * 12 * 64 + k * 64 + n] : 0.0f;
        o_tb1[t] = (_Float16)v;
    }
    if (t < 3 * 64 * 64) {               // [blk][n=64][k=64]
        int blk = t / (64 * 64);
        int rc  = t % (64 * 64);
        int n   = rc >> 6, k = rc & 63;
        size_t s = (size_t)blk * 64 * 64 + (size_t)k * 64 + n;
        o_tb2[t] = (_Float16)Wtb2[s];
        o_ega[t] = (_Float16)Wega[s];
        o_egg[t] = (_Float16)Wegg[s];
        o_ata[t] = (_Float16)Wata[s];
        o_atg[t] = (_Float16)Watg[s];
    }
}

// ---------------------------------------------------------------------------
// K1: initial edge features: edge_feat = rbf(r) @ W_edge + b_edge.
__global__ __launch_bounds__(256) void rbf_kernel(
    const float* __restrict__ edge_attr, const float* __restrict__ W_edge,
    const float* __restrict__ b_edge, float* __restrict__ edge_feat)
{
    __shared__ float sW[N_RADIAL * UNITS];
    __shared__ float sR[4][N_RADIAL];
    int tid = threadIdx.x;
    for (int i = tid; i < N_RADIAL * UNITS; i += 256) sW[i] = W_edge[i];

    int le = tid >> 6;                 // local edge 0..3
    int u  = tid & 63;
    size_t e = (size_t)blockIdx.x * 4 + le;
    float r  = (e < N_EDGES) ? edge_attr[e] : 1.0f;
    float rs = r < 1e-5f ? 1e-5f : r;
    if (u < N_RADIAL)
        sR[le][u] = __sinf((float)(u + 1) * PI_F * r * INV_CUTOFF) / rs;
    __syncthreads();

    float acc = b_edge[u];
#pragma unroll 4
    for (int n = 0; n < N_RADIAL; ++n) acc += sR[le][n] * sW[n * UNITS + u];
    if (e < N_EDGES) edge_feat[e * UNITS + u] = acc;
}

// ---------------------------------------------------------------------------
// K2: node features init: g = embedding[x].
__global__ __launch_bounds__(256) void init_g_kernel(
    const int* __restrict__ x, const float* __restrict__ emb, float* __restrict__ g)
{
    int i = blockIdx.x * 256 + threadIdx.x;
    if (i < N_NODES * UNITS) {
        int v = i >> 6, u = i & 63;
        g[i] = emb[(size_t)x[v] * UNITS + u];
    }
}

// ---------------------------------------------------------------------------
// K3: triplet kernel (per message block):
//   tb = basis[1M,12->32];  h = sigmoid(tb@W1);  w = sigmoid(h@W2);
//   atomicAdd(agg_e[tb_edge], w * edge_feat[tb_nbr])
__global__ __launch_bounds__(256) void trip_kernel(
    const float* __restrict__ angle, const float* __restrict__ edge_attr,
    const int* __restrict__ tbi,
    const _Float16* __restrict__ W1T, const _Float16* __restrict__ W2T,
    const float* __restrict__ b1, const float* __restrict__ b2,
    const float* __restrict__ edge_feat, float* __restrict__ agg_e)
{
    __shared__ _Float16 sW1T[64 * 32];      // [n][k] 4 KB
    __shared__ _Float16 sW2T[64 * 64];      // [n][k] 8 KB
    __shared__ _Float16 sTB[128 * 32];      // 8 KB
    __shared__ _Float16 sH[128 * UNITS];    // 16 KB
    __shared__ int sNbr[128];
    __shared__ int sEidx[128];

    const int tid = threadIdx.x;
    for (int i = tid; i < 64 * 32 / 8; i += 256) ((v8h*)sW1T)[i] = ((const v8h*)W1T)[i];
    for (int i = tid; i < 64 * 64 / 8; i += 256) ((v8h*)sW2T)[i] = ((const v8h*)W2T)[i];

    const int tbase = blockIdx.x * 128;
    if (tid < 128) {
        int t = tbase + tid;
        bool valid = t < N_TRIPLETS;
        int eix = 0, nbr = 0;
        float ang = 0.0f, r2 = 0.0f;
        if (valid) {
            eix = tbi[2 * t];
            nbr = tbi[2 * t + 1];
            ang = angle[t];
            r2  = edge_attr[nbr];
        }
        sNbr[tid]  = nbr;
        sEidx[tid] = eix;
        float r2s = r2 < 1e-5f ? 1e-5f : r2;
        float c[4];
#pragma unroll
        for (int l = 0; l < 4; ++l) c[l] = __cosf((float)l * ang);
        _Float16 row[12];
#pragma unroll
        for (int n = 0; n < 3; ++n) {
            float s = __sinf((float)(n + 1) * PI_F * r2 * INV_CUTOFF) / r2s;
            if (!valid) s = 0.0f;
#pragma unroll
            for (int l = 0; l < 4; ++l) row[n * 4 + l] = (_Float16)(s * c[l]);
        }
        v4h* prow = (v4h*)&sTB[tid * 32];
#pragma unroll
        for (int q = 0; q < 3; ++q)
            prow[q] = (v4h){row[4*q], row[4*q+1], row[4*q+2], row[4*q+3]};
        v4h z = {};
#pragma unroll
        for (int q = 3; q < 8; ++q) prow[q] = z;
    }
    __syncthreads();

    const int wave = tid >> 5;
    const int lane = tid & 31;
    const int rowbase = wave * 16;

    // mm1: [16x32] x [32x64], K padded (cols 12..31 of A are zero)
    v16h a = load_a_frag(&sTB[rowbase * 32], 32);
#pragma unroll
    for (int nt = 0; nt < 4; ++nt) {
        v16h b = load_bT_frag(&sW1T[(nt * 16) * 32], 32);
        v8f acc = {};
        acc = wmma_f16(a, b, acc);
        int n    = nt * 16 + (lane & 15);
        int moff = (lane >> 4) << 3;
        float bias = b1[n];
#pragma unroll
        for (int r = 0; r < 8; ++r) {
            int m = rowbase + moff + r;
            sH[m * UNITS + n] = (_Float16)sgm(acc[r] + bias);
        }
    }
    __syncthreads();

    // mm2: [16x64] x [64x64], then fused gather/scale/atomic-scatter
    v16h a0 = load_a_frag(&sH[rowbase * UNITS], UNITS);
    v16h a1 = load_a_frag(&sH[rowbase * UNITS + 32], UNITS);
#pragma unroll
    for (int nt = 0; nt < 4; ++nt) {
        v16h bk0 = load_bT_frag(&sW2T[(nt * 16) * 64], 64);
        v16h bk1 = load_bT_frag(&sW2T[(nt * 16) * 64 + 32], 64);
        v8f acc = {};
        acc = wmma_f16(a0, bk0, acc);
        acc = wmma_f16(a1, bk1, acc);
        int n    = nt * 16 + (lane & 15);
        int moff = (lane >> 4) << 3;
        float bias = b2[n];
#pragma unroll
        for (int r = 0; r < 8; ++r) {
            int ml = rowbase + moff + r;
            int t  = tbase + ml;
            if (t < N_TRIPLETS) {
                float w   = sgm(acc[r] + bias);
                int nbr   = sNbr[ml];
                int eix   = sEidx[ml];
                float msg = w * edge_feat[(size_t)nbr * UNITS + n];
                atomAddF(&agg_e[(size_t)eix * UNITS + n], msg);
            }
        }
    }
}

// ---------------------------------------------------------------------------
// K4: edge update:
//   upd = silu(agg_e@Wa+ba) * sigmoid(agg_e@Wg+bg); edge_feat += upd;
//   atomicAdd(agg_n[dst], edge_feat)
__global__ __launch_bounds__(256) void edge_kernel(
    const float* __restrict__ agg_e,
    const _Float16* __restrict__ WaT, const _Float16* __restrict__ WgT,
    const float* __restrict__ ba, const float* __restrict__ bg,
    const int* __restrict__ dst,
    float* __restrict__ edge_feat, float* __restrict__ agg_n)
{
    __shared__ _Float16 sWaT[64 * 64];
    __shared__ _Float16 sWgT[64 * 64];
    __shared__ _Float16 sX[128 * UNITS];
    __shared__ int sDst[128];

    int tid = threadIdx.x;
    for (int i = tid; i < 64 * 64 / 8; i += 256) {
        ((v8h*)sWaT)[i] = ((const v8h*)WaT)[i];
        ((v8h*)sWgT)[i] = ((const v8h*)WgT)[i];
    }
    size_t e0 = (size_t)blockIdx.x * 128;
    const float4* src = (const float4*)(agg_e + e0 * UNITS);
    v4h* dst4 = (v4h*)sX;
    for (int i = tid; i < 128 * UNITS / 4; i += 256) {
        float4 v = src[i];
        dst4[i] = (v4h){(_Float16)v.x, (_Float16)v.y, (_Float16)v.z, (_Float16)v.w};
    }
    if (tid < 128) sDst[tid] = dst[e0 + tid];
    __syncthreads();

    int wave = tid >> 5, lane = tid & 31;
    int rowbase = wave * 16;
    v16h a0 = load_a_frag(&sX[rowbase * UNITS], UNITS);
    v16h a1 = load_a_frag(&sX[rowbase * UNITS + 32], UNITS);
#pragma unroll
    for (int nt = 0; nt < 4; ++nt) {
        v16h ba0 = load_bT_frag(&sWaT[(nt * 16) * 64], 64);
        v16h ba1 = load_bT_frag(&sWaT[(nt * 16) * 64 + 32], 64);
        v16h bg0 = load_bT_frag(&sWgT[(nt * 16) * 64], 64);
        v16h bg1 = load_bT_frag(&sWgT[(nt * 16) * 64 + 32], 64);
        v8f aa = {}, ag = {};
        aa = wmma_f16(a0, ba0, aa);
        aa = wmma_f16(a1, ba1, aa);
        ag = wmma_f16(a0, bg0, ag);
        ag = wmma_f16(a1, bg1, ag);
        int n    = nt * 16 + (lane & 15);
        int moff = (lane >> 4) << 3;
        float bia = ba[n], big = bg[n];
#pragma unroll
        for (int r = 0; r < 8; ++r) {
            int ml   = rowbase + moff + r;
            size_t e = e0 + ml;
            float upd = silu_f(aa[r] + bia) * sgm(ag[r] + big);
            float ef  = edge_feat[e * UNITS + n] + upd;
            edge_feat[e * UNITS + n] = ef;
            atomAddF(&agg_n[(size_t)sDst[ml] * UNITS + n], ef);
        }
    }
}

// ---------------------------------------------------------------------------
// K5: node update: g += silu(agg_n@Wa+ba) * sigmoid(agg_n@Wg+bg)
__global__ __launch_bounds__(256) void node_kernel(
    const float* __restrict__ agg_n,
    const _Float16* __restrict__ WaT, const _Float16* __restrict__ WgT,
    const float* __restrict__ ba, const float* __restrict__ bg,
    float* __restrict__ g)
{
    __shared__ _Float16 sWaT[64 * 64];
    __shared__ _Float16 sWgT[64 * 64];
    __shared__ _Float16 sX[128 * UNITS];

    int tid = threadIdx.x;
    for (int i = tid; i < 64 * 64 / 8; i += 256) {
        ((v8h*)sWaT)[i] = ((const v8h*)WaT)[i];
        ((v8h*)sWgT)[i] = ((const v8h*)WgT)[i];
    }
    size_t v0 = (size_t)blockIdx.x * 128;
    const float4* src = (const float4*)(agg_n + v0 * UNITS);
    v4h* dst4 = (v4h*)sX;
    for (int i = tid; i < 128 * UNITS / 4; i += 256) {
        size_t v = v0 + (size_t)(i >> 4);
        v4h h = {};
        if (v < N_NODES) {
            float4 t = src[i];
            h = (v4h){(_Float16)t.x, (_Float16)t.y, (_Float16)t.z, (_Float16)t.w};
        }
        dst4[i] = h;
    }
    __syncthreads();

    int wave = tid >> 5, lane = tid & 31;
    int rowbase = wave * 16;
    v16h a0 = load_a_frag(&sX[rowbase * UNITS], UNITS);
    v16h a1 = load_a_frag(&sX[rowbase * UNITS + 32], UNITS);
#pragma unroll
    for (int nt = 0; nt < 4; ++nt) {
        v16h ba0 = load_bT_frag(&sWaT[(nt * 16) * 64], 64);
        v16h ba1 = load_bT_frag(&sWaT[(nt * 16) * 64 + 32], 64);
        v16h bg0 = load_bT_frag(&sWgT[(nt * 16) * 64], 64);
        v16h bg1 = load_bT_frag(&sWgT[(nt * 16) * 64 + 32], 64);
        v8f aa = {}, ag = {};
        aa = wmma_f16(a0, ba0, aa);
        aa = wmma_f16(a1, ba1, aa);
        ag = wmma_f16(a0, bg0, ag);
        ag = wmma_f16(a1, bg1, ag);
        int n    = nt * 16 + (lane & 15);
        int moff = (lane >> 4) << 3;
        float bia = ba[n], big = bg[n];
#pragma unroll
        for (int r = 0; r < 8; ++r) {
            size_t v = v0 + (size_t)(rowbase + moff + r);
            if (v < N_NODES) {
                float upd = silu_f(aa[r] + bia) * sgm(ag[r] + big);
                g[v * UNITS + n] += upd;
            }
        }
    }
}

// ---------------------------------------------------------------------------
// K6: readout per node: w = sigmoid(g.Wro_w+b); h = silu(g@Wro_h+b);
//     num[batch] += w*h ; den[batch] += w
__global__ __launch_bounds__(64) void readout_kernel(
    const float* __restrict__ g,
    const float* __restrict__ Ww, const float* __restrict__ bw,
    const float* __restrict__ Wh, const float* __restrict__ bh,
    const int* __restrict__ batch,
    float* __restrict__ num, float* __restrict__ den)
{
    __shared__ float sg[UNITS];
    int v = blockIdx.x;
    int u = threadIdx.x;
    sg[u] = g[(size_t)v * UNITS + u];
    __syncthreads();

    float wdot = bw[0];
#pragma unroll 8
    for (int k = 0; k < UNITS; ++k) wdot += sg[k] * Ww[k];
    float w = sgm(wdot);

    float hacc = bh[u];
#pragma unroll 8
    for (int k = 0; k < UNITS; ++k) hacc += sg[k] * Wh[k * UNITS + u];
    float h = silu_f(hacc);

    int b = batch[v];
    atomAddF(&num[(size_t)b * UNITS + u], w * h);
    if (u == 0) atomAddF(&den[b], w);
}

// ---------------------------------------------------------------------------
// K7: final MLP per graph: out = silu(pool@W_f1+b_f1)@W_f2+b_f2
__global__ __launch_bounds__(64) void final_kernel(
    const float* __restrict__ num, const float* __restrict__ den,
    const float* __restrict__ W1, const float* __restrict__ b1,
    const float* __restrict__ W2, const float* __restrict__ b2,
    float* __restrict__ out)
{
    __shared__ float sp[UNITS];
    __shared__ float st[UNITS];
    int gi = blockIdx.x;
    int u  = threadIdx.x;
    float d = den[gi] + 1e-8f;
    sp[u] = num[(size_t)gi * UNITS + u] / d;
    __syncthreads();
    float acc = b1[u];
#pragma unroll 8
    for (int k = 0; k < UNITS; ++k) acc += sp[k] * W1[k * UNITS + u];
    st[u] = silu_f(acc);
    __syncthreads();
    if (u == 0) {
        float o = b2[0];
        for (int k = 0; k < UNITS; ++k) o += st[k] * W2[k];
        out[gi] = o;
    }
}

// ---------------------------------------------------------------------------
extern "C" void kernel_launch(void* const* d_in, const int* in_sizes, int n_in,
                              void* d_out, int out_size, void* d_ws, size_t ws_size,
                              hipStream_t stream) {
    (void)in_sizes; (void)n_in; (void)out_size; (void)ws_size;
    const int*   x      = (const int*)  d_in[0];
    const int*   eindex = (const int*)  d_in[1];
    const float* eattr  = (const float*)d_in[2];
    const float* angle  = (const float*)d_in[3];
    const int*   tbi    = (const int*)  d_in[4];
    const int*   batch  = (const int*)  d_in[5];
    const float* emb    = (const float*)d_in[6];
    const float* W_edge = (const float*)d_in[7];
    const float* b_edge = (const float*)d_in[8];
    const float* W_tb1  = (const float*)d_in[9];
    const float* b_tb1  = (const float*)d_in[10];
    const float* W_tb2  = (const float*)d_in[11];
    const float* b_tb2  = (const float*)d_in[12];
    const float* W_eg_a = (const float*)d_in[13];
    const float* b_eg_a = (const float*)d_in[14];
    const float* W_eg_g = (const float*)d_in[15];
    const float* b_eg_g = (const float*)d_in[16];
    const float* W_at_a = (const float*)d_in[17];
    const float* b_at_a = (const float*)d_in[18];
    const float* W_at_g = (const float*)d_in[19];
    const float* b_at_g = (const float*)d_in[20];
    const float* W_ro_w = (const float*)d_in[21];
    const float* b_ro_w = (const float*)d_in[22];
    const float* W_ro_h = (const float*)d_in[23];
    const float* b_ro_h = (const float*)d_in[24];
    const float* W_f1   = (const float*)d_in[25];
    const float* b_f1   = (const float*)d_in[26];
    const float* W_f2   = (const float*)d_in[27];
    const float* b_f2   = (const float*)d_in[28];
    float* out = (float*)d_out;

    // workspace carving (256-byte aligned blocks)
    char* ws = (char*)d_ws;
    size_t off = 0;
    auto carve = [&](size_t bytes) -> char* {
        char* p = ws + off;
        off = (off + bytes + 255) & ~(size_t)255;
        return p;
    };
    float* edge_feat = (float*)carve((size_t)N_EDGES * UNITS * 4);
    float* agg_e     = (float*)carve((size_t)N_EDGES * UNITS * 4);
    float* agg_n     = (float*)carve((size_t)N_NODES * UNITS * 4);
    float* g         = (float*)carve((size_t)N_NODES * UNITS * 4);
    float* num       = (float*)carve((size_t)N_GRAPHS * UNITS * 4);
    float* den       = (float*)carve((size_t)N_GRAPHS * 4);
    _Float16* w16_tb1 = (_Float16*)carve(3 * 64 * 32 * 2);   // [n][k] transposed
    _Float16* w16_tb2 = (_Float16*)carve(3 * 64 * 64 * 2);
    _Float16* w16_ega = (_Float16*)carve(3 * 64 * 64 * 2);
    _Float16* w16_egg = (_Float16*)carve(3 * 64 * 64 * 2);
    _Float16* w16_ata = (_Float16*)carve(3 * 64 * 64 * 2);
    _Float16* w16_atg = (_Float16*)carve(3 * 64 * 64 * 2);

    const int* dst = eindex + N_EDGES;   // edge_index[1]

    // weight conversion (f32 -> f16 transposed, W_tb1 padded K 12->32)
    convert_weights_kernel<<<(3 * 64 * 64 + 255) / 256, 256, 0, stream>>>(
        W_tb1, W_tb2, W_eg_a, W_eg_g, W_at_a, W_at_g,
        w16_tb1, w16_tb2, w16_ega, w16_egg, w16_ata, w16_atg);

    // edge_feat init + g init
    rbf_kernel<<<N_EDGES / 4, 256, 0, stream>>>(eattr, W_edge, b_edge, edge_feat);
    init_g_kernel<<<(N_NODES * UNITS + 255) / 256, 256, 0, stream>>>(x, emb, g);

    hipMemsetAsync(num, 0, (size_t)N_GRAPHS * UNITS * 4, stream);
    hipMemsetAsync(den, 0, (size_t)N_GRAPHS * 4, stream);

    const int trip_blocks = (N_TRIPLETS + 127) / 128;
    const int edge_blocks = N_EDGES / 128;
    const int node_blocks = (N_NODES + 127) / 128;

    for (int i = 0; i < 3; ++i) {
        hipMemsetAsync(agg_e, 0, (size_t)N_EDGES * UNITS * 4, stream);
        hipMemsetAsync(agg_n, 0, (size_t)N_NODES * UNITS * 4, stream);

        trip_kernel<<<trip_blocks, 256, 0, stream>>>(
            angle, eattr, tbi,
            w16_tb1 + (size_t)i * 64 * 32, w16_tb2 + (size_t)i * 64 * 64,
            b_tb1 + i * UNITS, b_tb2 + i * UNITS,
            edge_feat, agg_e);

        edge_kernel<<<edge_blocks, 256, 0, stream>>>(
            agg_e,
            w16_ega + (size_t)i * 64 * 64, w16_egg + (size_t)i * 64 * 64,
            b_eg_a + i * UNITS, b_eg_g + i * UNITS,
            dst, edge_feat, agg_n);

        node_kernel<<<node_blocks, 256, 0, stream>>>(
            agg_n,
            w16_ata + (size_t)i * 64 * 64, w16_atg + (size_t)i * 64 * 64,
            b_at_a + i * UNITS, b_at_g + i * UNITS,
            g);
    }

    readout_kernel<<<N_NODES, 64, 0, stream>>>(
        g, W_ro_w, b_ro_w, W_ro_h, b_ro_h, batch, num, den);

    final_kernel<<<N_GRAPHS, 64, 0, stream>>>(
        num, den, W_f1, b_f1, W_f2, b_f2, out);
}
